// Attention_2851858284976
// MI455X (gfx1250) — matrix operand order repaired and checked by reading, not compiled, not run on hardware
//
#include <hip/hip_runtime.h>
#include <hip/hip_bf16.h>

typedef _Float16 f16;
typedef __attribute__((ext_vector_type(16))) _Float16 v16h;
typedef __attribute__((ext_vector_type(8)))  float    v8f;

union Frag { v16h v; uint4 q[2]; };

// A-matrix fragment (16x32, f16), ISA 7.12.2: lanes 0-15 hold M=lane with
// K = {0..7, 16..23}; lanes 16-31 hold same M with K offset +8.
__device__ __forceinline__ v16h load_a_frag(const f16* p, int ld) {
    const int lane = threadIdx.x & 31;
    const int row  = lane & 15;
    const int k0   = (lane & 16) ? 8 : 0;
    const f16* q = p + row * ld + k0;
    Frag f;
    f.q[0] = *(const uint4*)(q);
    f.q[1] = *(const uint4*)(q + 16);
    return f.v;
}

// B-matrix fragment (32x16, f16): lane L<16 holds N=L, K = {0..7, 16..23} (from a
// row-major [N][K] tile); lanes 16-31 hold N=L-16 with K offset +8.
__device__ __forceinline__ v16h load_b_frag(const f16* p, int ld) {
    const int lane = threadIdx.x & 31;
    const int row  = lane & 15;
    const int k0   = (lane & 16) ? 8 : 0;
    const f16* q = p + row * ld + k0;
    Frag f;
    f.q[0] = *(const uint4*)(q);
    f.q[1] = *(const uint4*)(q + 16);
    return f.v;
}

__device__ __forceinline__ v8f wmma_f16(v16h a, v16h b, v8f c) {
    return __builtin_amdgcn_wmma_f32_16x16x32_f16(false, a, false, b, (short)0, c,
                                                  false, false);
}

// ---- CDNA5 async global->LDS copy (ASYNCcnt-tracked, VGLOBAL op 98) ----
// LDS byte address = low 32 bits of the generic pointer ({SHARED_BASE, addr[31:0]}).
__device__ __forceinline__ void async_g2l_b128(void* lds_dst, const void* gsrc) {
    unsigned loff = (unsigned)(unsigned long long)lds_dst;
    asm volatile("global_load_async_to_lds_b128 %0, %1, off"
                 :: "v"(loff), "v"(gsrc)
                 : "memory");
}
__device__ __forceinline__ void wait_async_le8() {
    asm volatile("s_wait_asynccnt 0x8" ::: "memory");
}
__device__ __forceinline__ void wait_async_0() {
    asm volatile("s_wait_asynccnt 0x0" ::: "memory");
}

// fp32 -> f16 pack pass (run once per tensor; all sizes divisible by 8)
__global__ __launch_bounds__(256) void cvt_f32_to_f16(const float* __restrict__ src,
                                                      f16* __restrict__ dst, int n) {
    const int i = (blockIdx.x * 256 + threadIdx.x) * 8;
    if (i + 8 <= n) {
        float4 a = ((const float4*)(src + i))[0];
        float4 b = ((const float4*)(src + i))[1];
        __align__(16) f16 t[8] = {(f16)a.x, (f16)a.y, (f16)a.z, (f16)a.w,
                                  (f16)b.x, (f16)b.y, (f16)b.z, (f16)b.w};
        *(uint4*)(dst + i) = *(const uint4*)t;
    }
}

// C[M,N] = A[M,K] * W[N,K]^T (+bias). A, W already f16. Block tile 128x128,
// K-step 64, double-buffered async LDS staging, 8 waves as 2x4 of 64x32 tiles.
template <bool OUT_HALF, bool BIAS>
__global__ __launch_bounds__(256) void gemm_wmma(const f16* __restrict__ A,
                                                 const f16* __restrict__ Bw,
                                                 const float* __restrict__ bias,
                                                 void* __restrict__ Cp,
                                                 int M, int N, int K) {
    constexpr int BM = 128, BN = 128, BK = 64;
    __shared__ __align__(16) f16 sA[2][BM * BK];   // 2 x 16 KB
    __shared__ __align__(16) f16 sB[2][BN * BK];   // 2 x 16 KB
    const int tid  = threadIdx.x;
    const int wid  = tid >> 5;
    const int lane = tid & 31;
    const int m0 = blockIdx.x * BM;
    const int n0 = blockIdx.y * BN;
    const int wm = wid & 1;   // 0..1 -> 64 rows each
    const int wn = wid >> 1;  // 0..3 -> 32 cols each

    const int srow = tid >> 1;           // 0..127
    const int sc0  = (tid & 1) * 32;     // 2 x 32-element chunks per row

    auto stage = [&](int k0, int bufi) {
        const f16* ga = A  + (size_t)(m0 + srow) * K + k0 + sc0;
        const f16* gb = Bw + (size_t)(n0 + srow) * K + k0 + sc0;
        f16* la = sA[bufi] + srow * BK + sc0;
        f16* lb = sB[bufi] + srow * BK + sc0;
#pragma unroll
        for (int i = 0; i < 4; i++) {    // 4 x 16B each for A and B
            async_g2l_b128(la + 8 * i, ga + 8 * i);
            async_g2l_b128(lb + 8 * i, gb + 8 * i);
        }
    };

    v8f acc[4][2] = {};
    const int S = K / BK;
    stage(0, 0);
    for (int s = 0; s < S; s++) {
        if (s + 1 < S) {
            stage((s + 1) * BK, (s + 1) & 1);  // prefetch next tile (8 copies)
            wait_async_le8();                   // in-order: stage s has landed
        } else {
            wait_async_0();
        }
        __syncthreads();
        const f16* cA = sA[s & 1];
        const f16* cB = sB[s & 1];
#pragma unroll
        for (int kk = 0; kk < BK; kk += 32) {
            v16h a[4], b[2];
#pragma unroll
            for (int i = 0; i < 4; i++)
                a[i] = load_a_frag(cA + (wm * 64 + i * 16) * BK + kk, BK);
#pragma unroll
            for (int j = 0; j < 2; j++)
                b[j] = load_b_frag(cB + (wn * 32 + j * 16) * BK + kk, BK);
#pragma unroll
            for (int i = 0; i < 4; i++)
#pragma unroll
                for (int j = 0; j < 2; j++)
                    acc[i][j] = wmma_f16(a[i], b[j], acc[i][j]);
        }
        __syncthreads();
    }
    // C/D layout: VGPR r -> M = r (+8 for lanes 16-31), N = lane&15
    const int rbase = (lane & 16) ? 8 : 0;
    const int cl    = lane & 15;
#pragma unroll
    for (int i = 0; i < 4; i++) {
#pragma unroll
        for (int j = 0; j < 2; j++) {
            const int mb = m0 + wm * 64 + i * 16 + rbase;
            const int nb = n0 + wn * 32 + j * 16 + cl;
#pragma unroll
            for (int r = 0; r < 8; r++) {
                const size_t idx = (size_t)(mb + r) * N + nb;
                if (OUT_HALF)
                    ((f16*)Cp)[idx] = (f16)acc[i][j][r];
                else
                    ((float*)Cp)[idx] = acc[i][j][r] + (BIAS ? bias[nb] : 0.f);
            }
        }
    }
}

// One workgroup per (batch, head, 32-query block). qkv laid out so q/k/v heads
// are contiguous slices (matches reference's raw reshape). 8 waves.
__global__ __launch_bounds__(256) void attn_wmma(const f16* __restrict__ qkv,
                                                 f16* __restrict__ o) {
    constexpr int N = 1024, HD = 64, QB = 32;
    __shared__ __align__(16) f16 sQ[QB * HD];     // 4 KB
    __shared__ __align__(16) f16 sSP[QB * N];     // 64 KB: scores, then probs (in place)
    __shared__ __align__(16) f16 sVt[HD * 128];   // 16 KB: V^T chunk [d][key]

    const int tid  = threadIdx.x;
    const int lane = tid & 31;
    const int wid  = tid >> 5;
    const int qb = blockIdx.x & 31;
    const int h  = (blockIdx.x >> 5) & 15;
    const int b  = blockIdx.x >> 9;

    const size_t base = (size_t)b * 3145728;  // n * 3*dim per batch
    const f16* Qg = qkv + base + (size_t)h * 65536 + (size_t)qb * QB * HD;
    const f16* Kg = qkv + base + 1048576 + (size_t)h * 65536;
    const f16* Vg = qkv + base + 2097152 + (size_t)h * 65536;

    // stage Q block (32x64)
    ((uint4*)sQ)[tid] = ((const uint4*)Qg)[tid];
    __syncthreads();

    // ---- Phase A: S = scale * Q K^T (K fragments straight from global) ----
    v16h qa[2][2];
#pragma unroll
    for (int rt = 0; rt < 2; rt++)
#pragma unroll
        for (int kk = 0; kk < 2; kk++)
            qa[rt][kk] = load_a_frag(sQ + rt * 16 * HD + kk * 32, HD);

    const float scale = 0.125f;  // head_dim^-0.5
    for (int c8 = 0; c8 < 8; c8++) {
        const int n0 = (wid * 8 + c8) * 16;
        v16h kb0 = load_b_frag(Kg + (size_t)n0 * HD + 0,  HD);
        v16h kb1 = load_b_frag(Kg + (size_t)n0 * HD + 32, HD);
#pragma unroll
        for (int rt = 0; rt < 2; rt++) {
            v8f acc = {};
            acc = wmma_f16(qa[rt][0], kb0, acc);
            acc = wmma_f16(qa[rt][1], kb1, acc);
            const int rowb = rt * 16 + ((lane & 16) ? 8 : 0);
            const int col  = n0 + (lane & 15);
#pragma unroll
            for (int r = 0; r < 8; r++)
                sSP[(rowb + r) * N + col] = (f16)(acc[r] * scale);
        }
    }
    __syncthreads();

    // ---- Phase B: row softmax, in place (each lane rewrites what it read) ----
    for (int rr = 0; rr < 4; rr++) {
        const int row = wid * 4 + rr;
        float vals[32];
        float mx = -3.4e38f;
#pragma unroll
        for (int j = 0; j < 32; j++) {
            vals[j] = (float)sSP[row * N + lane + j * 32];
            mx = fmaxf(mx, vals[j]);
        }
#pragma unroll
        for (int off = 16; off > 0; off >>= 1) mx = fmaxf(mx, __shfl_xor(mx, off, 32));
        float sum = 0.f;
#pragma unroll
        for (int j = 0; j < 32; j++) {
            float e = __expf(vals[j] - mx);
            vals[j] = e;
            sum += e;
        }
#pragma unroll
        for (int off = 16; off > 0; off >>= 1) sum += __shfl_xor(sum, off, 32);
        const float inv = 1.f / sum;
#pragma unroll
        for (int j = 0; j < 32; j++)
            sSP[row * N + lane + j * 32] = (f16)(vals[j] * inv);
    }
    __syncthreads();

    // ---- Phase C: O = P V. Stage V transposed so B-fragments are contiguous ----
    const int rt = wid & 1;   // query row tile
    const int dt = wid >> 1;  // d column tile (0..3)
    v8f oacc = {};
    for (int kc = 0; kc < 8; kc++) {
        const int key0 = kc * 128;
        {
            const int krow = tid >> 1;
            const int d0   = (tid & 1) * 32;
            const f16* src = Vg + (size_t)(key0 + krow) * HD + d0;
#pragma unroll
            for (int i = 0; i < 32; i++) sVt[(d0 + i) * 128 + krow] = src[i];
        }
        __syncthreads();
#pragma unroll
        for (int kk = 0; kk < 4; kk++) {
            v16h pa = load_a_frag(sSP + (rt * 16) * N + key0 + kk * 32, N);
            v16h vb = load_b_frag(sVt + (dt * 16) * 128 + kk * 32, 128);
            oacc = wmma_f16(pa, vb, oacc);
        }
        __syncthreads();
    }
    {   // o[b][n][h*64+d]  (matches reference transpose+reshape)
        const int rbase = (lane & 16) ? 8 : 0;
        const int nnb   = qb * QB + rt * 16 + rbase;
        const int col   = h * HD + dt * 16 + (lane & 15);
#pragma unroll
        for (int r = 0; r < 8; r++)
            o[(size_t)(b * 1024 + nnb + r) * 1024 + col] = (f16)oacc[r];
    }
}

extern "C" void kernel_launch(void* const* d_in, const int* in_sizes, int n_in,
                              void* d_out, int out_size, void* d_ws, size_t ws_size,
                              hipStream_t stream) {
    const float* x     = (const float*)d_in[0];
    const float* w_qkv = (const float*)d_in[1];
    const float* w_out = (const float*)d_in[2];
    const float* b_out = (const float*)d_in[3];
    float* out = (float*)d_out;

    char* ws = (char*)d_ws;
    f16* qkv = (f16*)(ws);                              // 48 MB
    f16* oo  = (f16*)(ws + 48ull * 1024 * 1024);        // 16 MB
    f16* xh  = (f16*)(ws + 64ull * 1024 * 1024);        // 16 MB
    f16* wqh = (f16*)(ws + 80ull * 1024 * 1024);        // 6 MB
    f16* woh = (f16*)(ws + 86ull * 1024 * 1024);        // 2 MB

    dim3 blk(256, 1, 1);
    // pre-convert fp32 -> f16 once
    cvt_f32_to_f16<<<dim3(8 * 1024 * 1024 / (8 * 256)), blk, 0, stream>>>(x, xh, 8 * 1024 * 1024);
    cvt_f32_to_f16<<<dim3(3 * 1024 * 1024 / (8 * 256)), blk, 0, stream>>>(w_qkv, wqh, 3 * 1024 * 1024);
    cvt_f32_to_f16<<<dim3(1024 * 1024 / (8 * 256)), blk, 0, stream>>>(w_out, woh, 1024 * 1024);

    // qkv = x @ w_qkv^T   (M=8192, N=3072, K=1024) -> f16
    gemm_wmma<true, false><<<dim3(8192 / 128, 3072 / 128), blk, 0, stream>>>(
        xh, wqh, nullptr, qkv, 8192, 3072, 1024);
    // attention per (b,h,32-row block)
    attn_wmma<<<dim3(4096), blk, 0, stream>>>(qkv, oo);
    // out = o @ w_out^T + b_out  (M=8192, N=1024, K=1024) -> fp32
    gemm_wmma<false, true><<<dim3(8192 / 128, 1024 / 128), blk, 0, stream>>>(
        oo, woh, b_out, out, 8192, 1024, 1024);
}
